// ActorNet_61229053772334
// MI455X (gfx1250) — compile-verified, matrix operand order checked by reading, not run on hardware
//
#include <hip/hip_runtime.h>
#include <hip/hip_bf16.h>
#include <stdint.h>

// ---------------------------------------------------------------------------
// MI455X (gfx1250) implementation of the MPNN ActorNet reference.
//
// msg GEMM: never materialize ew[E,64,64] (491MB). Fuse:
//   msg_tile[16,64] = (out_src ⊗ g)[16,4096] @ W2r[4096,64] + out_src@B2
// Within each K=32 WMMA step, the outer-product row index i is constant,
// so A = scalar * (one of two per-lane g fragments) -> pure v_pk_mul_f16
// fragment build, no LDS. Two M-tiles per wave share each B fragment
// (reuse_b hint on the second, identical WMMA) to halve L2 fragment traffic.
// ---------------------------------------------------------------------------

#define N_NODES 10000
#define N_EDGES 30000
#define T_ROWS  256
#define DIM     64

typedef __attribute__((ext_vector_type(16))) _Float16 v16h;
typedef __attribute__((ext_vector_type(8)))  _Float16 v8h;
typedef __attribute__((ext_vector_type(8)))  float    v8f;

// K index inside a 16x32 f16 A/B fragment: element e of lane's v16h.
// lanes 0-15: e0..7 -> K0..7, e8..15 -> K16..23 ; lanes 16-31: +8.
__device__ __forceinline__ int kmap(int e, int lane) {
    return (e & 7) + ((e >> 3) << 4) + (((lane >> 4) & 1) << 3);
}

__device__ __forceinline__ v16h cat8(v8h a, v8h b) {
    return __builtin_shufflevector(a, b, 0, 1, 2, 3, 4, 5, 6, 7,
                                   8, 9, 10, 11, 12, 13, 14, 15);
}

__device__ __forceinline__ float sigmf(float x) {
    return 1.0f / (1.0f + __expf(-x));
}

// ------------------------------- utility kernels ---------------------------

__global__ void zero_f32(float* __restrict__ p, int n) {
    int i = blockIdx.x * blockDim.x + threadIdx.x;
    if (i < n) p[i] = 0.0f;
}

__global__ void deg_kernel(const int* __restrict__ dst, float* __restrict__ deg, int e) {
    int i = blockIdx.x * blockDim.x + threadIdx.x;
    if (i < e) atomicAdd(&deg[dst[i]], 1.0f);
}

__global__ void invdeg_kernel(float* __restrict__ deg, int n) {
    int i = blockIdx.x * blockDim.x + threadIdx.x;
    if (i < n) deg[i] = 1.0f / fmaxf(deg[i], 1.0f);
}

// out = relu(x @ lin0_w^T + b)  ; node state (f32 + f16 copies)
__global__ void init_node(const float* __restrict__ x, const float* __restrict__ w,
                          const float* __restrict__ b,
                          float* __restrict__ nf, _Float16* __restrict__ nh) {
    int idx = blockIdx.x * blockDim.x + threadIdx.x;   // N*64 exact
    int n = idx >> 6, c = idx & 63;
    float v = b[c];
    #pragma unroll
    for (int k = 0; k < 3; ++k) v += x[n * 3 + k] * w[c * 3 + k];
    v = fmaxf(v, 0.0f);
    nf[idx] = v;
    nh[idx] = (_Float16)v;
}

// g = relu(edge_attr @ W1^T + b1) in f16
__global__ void edge_g(const float* __restrict__ ea, const float* __restrict__ w1,
                       const float* __restrict__ b1, _Float16* __restrict__ g) {
    int idx = blockIdx.x * blockDim.x + threadIdx.x;   // E*64 exact
    int e = idx >> 6, c = idx & 63;
    float v = b1[c];
    #pragma unroll
    for (int k = 0; k < 7; ++k) v += ea[e * 7 + k] * w1[c * 7 + k];
    g[idx] = (_Float16)fmaxf(v, 0.0f);
}

// ----------------------- fragment-order weight packing ---------------------
// layout: frag[((kb*NT + nt)*32 + lane)*16 + t]  => one contiguous v16h/lane

__global__ void prep_w2r(const float* __restrict__ w2, _Float16* __restrict__ w2f) {
    int idx = blockIdx.x * blockDim.x + threadIdx.x;   // 128*4*32*16 = 262144
    int t = idx & 15, lane = (idx >> 4) & 31, nt = (idx >> 9) & 3, kb = idx >> 11;
    int kk = kb * 32 + kmap(t, lane);                  // kappa = i*64 + k
    int o  = nt * 16 + (lane & 15);
    w2f[idx] = (_Float16)w2[((kk >> 6) * 64 + o) * 64 + (kk & 63)];
}

__global__ void prep_b2(const float* __restrict__ b2, _Float16* __restrict__ b2f) {
    int idx = blockIdx.x * blockDim.x + threadIdx.x;   // 2*4*32*16 = 4096
    int t = idx & 15, lane = (idx >> 4) & 31, nt = (idx >> 9) & 3, kb = idx >> 11;
    int kk = kb * 32 + kmap(t, lane);                  // K = i  (0..63)
    int o  = nt * 16 + (lane & 15);
    b2f[idx] = (_Float16)b2[kk * 64 + o];
}

// gru W [192,64] -> frag order over K=64 (2 kb), 12 column tiles
__global__ void prep_gruw(const float* __restrict__ w, _Float16* __restrict__ wf) {
    int idx = blockIdx.x * blockDim.x + threadIdx.x;   // 2*12*32*16 = 12288
    int t = idx & 15, lane = (idx >> 4) & 31;
    int ct = (idx >> 9) % 12, kb = (idx >> 9) / 12;
    int kk = kb * 32 + kmap(t, lane);
    int j  = ct * 16 + (lane & 15);
    wf[idx] = (_Float16)w[j * 64 + kk];
}

// lin1_w [64,320] -> frag order over K=320 (10 kb), 4 column tiles
__global__ void prep_lin1(const float* __restrict__ w, _Float16* __restrict__ wf) {
    int idx = blockIdx.x * blockDim.x + threadIdx.x;   // 10*4*32*16 = 20480
    int t = idx & 15, lane = (idx >> 4) & 31;
    int ct = (idx >> 9) & 3, kb = idx >> 11;
    int kk = kb * 32 + kmap(t, lane);
    int c  = ct * 16 + (lane & 15);
    wf[idx] = (_Float16)w[c * 320 + kk];
}

// ------------------------- fused edge message GEMM -------------------------
// MT 16-edge tiles per wave; B fragments shared across tiles (reuse_b hint).
template <int MT>
__device__ __forceinline__ void msg_tiles(int e0, int lane,
                                          const _Float16* __restrict__ node_h,
                                          const _Float16* __restrict__ g_h,
                                          const _Float16* __restrict__ w2f,
                                          const _Float16* __restrict__ b2f,
                                          const int* __restrict__ src,
                                          const int* __restrict__ dst,
                                          float* __restrict__ agg) {
    const int m = lane & 15;
    const int hi8 = (lane >> 4) << 3;

    v8f acc[MT][4];
    #pragma unroll
    for (int mt = 0; mt < MT; ++mt)
        #pragma unroll
        for (int nt = 0; nt < 4; ++nt)
            #pragma unroll
            for (int j = 0; j < 8; ++j) acc[mt][nt][j] = 0.0f;

    // per-lane g fragments (constant across the entire K loop) + out row ptrs
    v16h Glo[MT], Ghi[MT];
    const _Float16* orow[MT];
    #pragma unroll
    for (int mt = 0; mt < MT; ++mt) {
        const _Float16* gr = g_h + (size_t)(e0 + mt * 16 + m) * 64;
        Glo[mt] = cat8(*(const v8h*)(gr + hi8),      *(const v8h*)(gr + 16 + hi8));
        Ghi[mt] = cat8(*(const v8h*)(gr + 32 + hi8), *(const v8h*)(gr + 48 + hi8));
        orow[mt] = node_h + (size_t)src[e0 + mt * 16 + m] * 64;
    }

    // main K loop: i = outer-product row (0..63); per i two K=32 steps
    for (int ib = 0; ib < 8; ++ib) {
        v8h O[MT];
        #pragma unroll
        for (int mt = 0; mt < MT; ++mt) O[mt] = *(const v8h*)(orow[mt] + ib * 8);
        #pragma unroll
        for (int j = 0; j < 8; ++j) {
            const int i = ib * 8 + j;
            v16h Al[MT], Ah[MT];
            #pragma unroll
            for (int mt = 0; mt < MT; ++mt) {
                _Float16 s = O[mt][j];
                #pragma unroll
                for (int t = 0; t < 16; ++t) {
                    Al[mt][t] = Glo[mt][t] * s;
                    Ah[mt][t] = Ghi[mt][t] * s;
                }
            }
            #pragma unroll
            for (int kh = 0; kh < 2; ++kh) {
                const int kb = 2 * i + kh;
                #pragma unroll
                for (int nt = 0; nt < 4; ++nt) {
                    v16h B = *(const v16h*)(w2f + (((kb * 4 + nt) * 32 + lane) << 4));
                    acc[0][nt] = __builtin_amdgcn_wmma_f32_16x16x32_f16(
                        false, kh ? Ah[0] : Al[0], false, B, (short)0,
                        acc[0][nt], false, false);
                    if constexpr (MT > 1) {
                        acc[1][nt] = __builtin_amdgcn_wmma_f32_16x16x32_f16(
                            false, kh ? Ah[1] : Al[1], false, B, (short)0,
                            acc[1][nt], false, true);
                    }
                }
            }
        }
    }

    // bias term: out_src @ B2 over K = i (64)
    v16h AoL[MT], AoH[MT];
    #pragma unroll
    for (int mt = 0; mt < MT; ++mt) {
        AoL[mt] = cat8(*(const v8h*)(orow[mt] + hi8),      *(const v8h*)(orow[mt] + 16 + hi8));
        AoH[mt] = cat8(*(const v8h*)(orow[mt] + 32 + hi8), *(const v8h*)(orow[mt] + 48 + hi8));
    }
    #pragma unroll
    for (int kh = 0; kh < 2; ++kh)
        #pragma unroll
        for (int nt = 0; nt < 4; ++nt) {
            v16h B = *(const v16h*)(b2f + (((kh * 4 + nt) * 32 + lane) << 4));
            acc[0][nt] = __builtin_amdgcn_wmma_f32_16x16x32_f16(
                false, kh ? AoH[0] : AoL[0], false, B, (short)0,
                acc[0][nt], false, false);
            if constexpr (MT > 1) {
                acc[1][nt] = __builtin_amdgcn_wmma_f32_16x16x32_f16(
                    false, kh ? AoH[1] : AoL[1], false, B, (short)0,
                    acc[1][nt], false, true);
            }
        }

    // scatter-add into destination nodes (mean scaling applied later)
    #pragma unroll
    for (int mt = 0; mt < MT; ++mt) {
        int d[8];
        #pragma unroll
        for (int r = 0; r < 8; ++r) d[r] = dst[e0 + mt * 16 + r + hi8];
        #pragma unroll
        for (int nt = 0; nt < 4; ++nt)
            #pragma unroll
            for (int r = 0; r < 8; ++r)
                atomicAdd(&agg[(size_t)d[r] * 64 + nt * 16 + m], acc[mt][nt][r]);
    }
}

__global__ __launch_bounds__(32)
void msg_kernel(const _Float16* __restrict__ node_h, const _Float16* __restrict__ g_h,
                const _Float16* __restrict__ w2f, const _Float16* __restrict__ b2f,
                const int* __restrict__ src, const int* __restrict__ dst,
                float* __restrict__ agg) {
    const int lane = threadIdx.x;
    const int e0 = blockIdx.x * 32;
    if (e0 + 32 <= N_EDGES) {            // wave-uniform branch: EXEC stays all-1s
        msg_tiles<2>(e0, lane, node_h, g_h, w2f, b2f, src, dst, agg);
    } else {
        msg_tiles<1>(e0, lane, node_h, g_h, w2f, b2f, src, dst, agg);
    }
}

// m = relu(agg*inv_deg + out@conv_root + conv_bias)  -> m_f16 for GRU GEMM
__global__ void m_kernel(const float* __restrict__ agg, const float* __restrict__ invd,
                         const float* __restrict__ nf, const float* __restrict__ root,
                         const float* __restrict__ bias, _Float16* __restrict__ mh) {
    int idx = blockIdx.x * blockDim.x + threadIdx.x;   // N*64 exact
    int n = idx >> 6, c = idx & 63;
    float v = agg[idx] * invd[n] + bias[c];
    #pragma unroll 8
    for (int k = 0; k < 64; ++k) v += nf[n * 64 + k] * root[k * 64 + c];
    mh[idx] = (_Float16)fmaxf(v, 0.0f);
}

// GRU: gi = m@Wih^T, gh = h@Whh^T via WMMA, then gates; state updated in place
__global__ __launch_bounds__(32)
void gru_kernel(const _Float16* __restrict__ mh, const _Float16* __restrict__ wihf,
                const _Float16* __restrict__ whhf, const float* __restrict__ bih,
                const float* __restrict__ bhh,
                float* __restrict__ nf, _Float16* __restrict__ nh) {
    __shared__ float sGi[16 * 192];
    __shared__ float sGh[16 * 192];
    const int lane = threadIdx.x;
    const int n0 = blockIdx.x * 16;
    const int m = lane & 15;
    const int hi8 = (lane >> 4) << 3;

    const _Float16* mrow = mh + (size_t)(n0 + m) * 64;
    const _Float16* hrow = nh + (size_t)(n0 + m) * 64;
    v16h Am[2], Ah[2];
    #pragma unroll
    for (int kb = 0; kb < 2; ++kb) {
        Am[kb] = cat8(*(const v8h*)(mrow + kb * 32 + hi8),
                      *(const v8h*)(mrow + kb * 32 + 16 + hi8));
        Ah[kb] = cat8(*(const v8h*)(hrow + kb * 32 + hi8),
                      *(const v8h*)(hrow + kb * 32 + 16 + hi8));
    }

    for (int ct = 0; ct < 12; ++ct) {
        v8f ai, ah;
        #pragma unroll
        for (int j = 0; j < 8; ++j) { ai[j] = 0.0f; ah[j] = 0.0f; }
        #pragma unroll
        for (int kb = 0; kb < 2; ++kb) {
            v16h Bi = *(const v16h*)(wihf + (((kb * 12 + ct) * 32 + lane) << 4));
            v16h Bh = *(const v16h*)(whhf + (((kb * 12 + ct) * 32 + lane) << 4));
            ai = __builtin_amdgcn_wmma_f32_16x16x32_f16(
                false, Am[kb], false, Bi, (short)0, ai, false, false);
            ah = __builtin_amdgcn_wmma_f32_16x16x32_f16(
                false, Ah[kb], false, Bh, (short)0, ah, false, false);
        }
        #pragma unroll
        for (int r = 0; r < 8; ++r) {
            int row = r + hi8, col = ct * 16 + m;
            sGi[row * 192 + col] = ai[r];
            sGh[row * 192 + col] = ah[r];
        }
    }
    __syncthreads();

    for (int idx = lane; idx < 16 * 64; idx += 32) {
        int r = idx >> 6, c = idx & 63;
        int n = n0 + r;
        float ir = sGi[r * 192 + c]       + bih[c];
        float iz = sGi[r * 192 + 64 + c]  + bih[64 + c];
        float in_= sGi[r * 192 + 128 + c] + bih[128 + c];
        float hr = sGh[r * 192 + c]       + bhh[c];
        float hz = sGh[r * 192 + 64 + c]  + bhh[64 + c];
        float hn = sGh[r * 192 + 128 + c] + bhh[128 + c];
        float rr = sigmf(ir + hr);
        float zz = sigmf(iz + hz);
        float nn = tanhf(in_ + rr * hn);
        float hp = nf[n * 64 + c];
        float hv = (1.0f - zz) * nn + zz * hp;
        nf[n * 64 + c] = hv;
        nh[n * 64 + c] = (_Float16)hv;
    }
}

// -------------------- Set2Set pooling + mem-LSTM (one WG) ------------------
__global__ __launch_bounds__(256)
void set2set_kernel(const float* __restrict__ nf,
                    const float* __restrict__ wih, const float* __restrict__ whh,
                    const float* __restrict__ bih, const float* __restrict__ bhh,
                    const float* __restrict__ mwih, const float* __restrict__ mbih,
                    const float* __restrict__ mbhh,
                    float* __restrict__ e_arr, float* __restrict__ hx_ws,
                    float* __restrict__ out_tail) {
    __shared__ float sh[64], sc[64], qs[128], g[256], red[256], rpart[256];
    const int t = threadIdx.x;
    if (t < 64) { sh[t] = 0.0f; sc[t] = 0.0f; }
    if (t < 128) qs[t] = 0.0f;
    __syncthreads();

    for (int step = 0; step < 6; ++step) {
        float acc = bih[t] + bhh[t];
        for (int k = 0; k < 128; ++k) acc += wih[t * 128 + k] * qs[k];
        for (int k = 0; k < 64; ++k)  acc += whh[t * 64 + k] * sh[k];
        g[t] = acc;
        __syncthreads();
        if (t < 64) {
            float i = sigmf(g[t]);
            float f = sigmf(g[64 + t]);
            float gg = tanhf(g[128 + t]);
            float o = sigmf(g[192 + t]);
            float c = f * sc[t] + i * gg;
            sc[t] = c;
            sh[t] = o * tanhf(c);
        }
        __syncthreads();
        float lmax = -1e30f;
        for (int n = t; n < N_NODES; n += 256) {
            float d = 0.0f;
            for (int c = 0; c < 64; ++c) d += nf[n * 64 + c] * sh[c];
            e_arr[n] = d;
            lmax = fmaxf(lmax, d);
        }
        red[t] = lmax; __syncthreads();
        for (int s = 128; s > 0; s >>= 1) {
            if (t < s) red[t] = fmaxf(red[t], red[t + s]);
            __syncthreads();
        }
        float emax = red[0];
        __syncthreads();
        float lsum = 0.0f;
        for (int n = t; n < N_NODES; n += 256) {
            float ex = __expf(e_arr[n] - emax);
            e_arr[n] = ex;
            lsum += ex;
        }
        red[t] = lsum;
        __threadfence_block();
        __syncthreads();
        for (int s = 128; s > 0; s >>= 1) {
            if (t < s) red[t] += red[t + s];
            __syncthreads();
        }
        float inv = 1.0f / red[0];
        __syncthreads();
        int c = t & 63, grp = t >> 6;
        float racc = 0.0f;
        for (int n = grp; n < N_NODES; n += 4) racc += e_arr[n] * nf[n * 64 + c];
        rpart[t] = racc; __syncthreads();
        if (t < 64) {
            float r = (rpart[t] + rpart[64 + t] + rpart[128 + t] + rpart[192 + t]) * inv;
            qs[t] = sh[t];
            qs[64 + t] = r;
        }
        __syncthreads();
    }
    float acc = mbih[t] + mbhh[t];
    for (int k = 0; k < 128; ++k) acc += mwih[t * 128 + k] * qs[k];
    g[t] = acc;
    __syncthreads();
    if (t < 64) {
        float i = sigmf(g[t]);
        float gg = tanhf(g[128 + t]);
        float o = sigmf(g[192 + t]);
        float c = i * gg;                 // c0 = 0
        float hxv = o * tanhf(c);
        hx_ws[t] = hxv;
        out_tail[t] = hxv;                // hx at d_out[1536..1599]
        out_tail[64 + t] = c;             // cx at d_out[1600..1663]
    }
}

// --------------------------- final MLP (WMMA) ------------------------------
__global__ __launch_bounds__(32)
void final_kernel(const float* __restrict__ nf, const int* __restrict__ nr,
                  const float* __restrict__ hx, const _Float16* __restrict__ lin1f,
                  const float* __restrict__ lin1b, const float* __restrict__ lin2w,
                  const float* __restrict__ lin2b, float* __restrict__ logits) {
    __shared__ _Float16 sF[16 * 320];
    __shared__ float sH[16 * 64];
    const int lane = threadIdx.x;
    const int t0 = blockIdx.x * 16;

    for (int idx = lane; idx < 16 * 320; idx += 32) {
        int r = idx / 320, j = idx % 320;
        int t = t0 + r;
        float v;
        if (j < 256) {
            int f = j * 256 + t;                       // sel.view(4*DIM,-1).T
            v = nf[(size_t)nr[f >> 6] * 64 + (f & 63)];
        } else {
            v = hx[(t * 64 + (j - 256)) >> 8];         // repeat_interleave quirk
        }
        sF[r * 320 + j] = (_Float16)v;
    }
    __syncthreads();

    const int m = lane & 15;
    const int hi8 = (lane >> 4) << 3;
    for (int ct = 0; ct < 4; ++ct) {
        v8f acc;
        #pragma unroll
        for (int j = 0; j < 8; ++j) acc[j] = 0.0f;
        #pragma unroll
        for (int kb = 0; kb < 10; ++kb) {
            v16h A;
            #pragma unroll
            for (int t = 0; t < 16; ++t) {
                int kk = kb * 32 + (t & 7) + ((t >> 3) << 4) + hi8;
                A[t] = sF[m * 320 + kk];
            }
            v16h B = *(const v16h*)(lin1f + (((kb * 4 + ct) * 32 + lane) << 4));
            acc = __builtin_amdgcn_wmma_f32_16x16x32_f16(
                false, A, false, B, (short)0, acc, false, false);
        }
        #pragma unroll
        for (int r = 0; r < 8; ++r) {
            int row = r + hi8, col = ct * 16 + m;
            sH[row * 64 + col] = fmaxf(acc[r] + lin1b[col], 0.0f);
        }
    }
    __syncthreads();

    for (int idx = lane; idx < 16 * 6; idx += 32) {
        int r = idx / 6, a = idx % 6;
        float s = lin2b[a];
        for (int c = 0; c < 64; ++c) s += sH[r * 64 + c] * lin2w[a * 64 + c];
        logits[(t0 + r) * 6 + a] = s;
    }
}

// ------------------------------- launcher ----------------------------------

static inline void* carve(uintptr_t& p, size_t bytes) {
    void* r = (void*)p;
    p += (bytes + 255) & ~(size_t)255;
    return r;
}

extern "C" void kernel_launch(void* const* d_in, const int* in_sizes, int n_in,
                              void* d_out, int out_size, void* d_ws, size_t ws_size,
                              hipStream_t stream) {
    (void)in_sizes; (void)n_in; (void)out_size; (void)ws_size;
    const float* x         = (const float*)d_in[0];
    const float* edge_attr = (const float*)d_in[1];
    const int*   edge_idx  = (const int*)d_in[2];
    const int*   nonring   = (const int*)d_in[3];
    // d_in[4] = batch (all zeros, B=1) -> unused
    const float* lin0_w  = (const float*)d_in[5];
    const float* lin0_b  = (const float*)d_in[6];
    const float* emlp_w1 = (const float*)d_in[7];
    const float* emlp_b1 = (const float*)d_in[8];
    const float* emlp_w2 = (const float*)d_in[9];
    const float* emlp_b2 = (const float*)d_in[10];
    const float* conv_root = (const float*)d_in[11];
    const float* conv_bias = (const float*)d_in[12];
    const float* gru_wih = (const float*)d_in[13];
    const float* gru_whh = (const float*)d_in[14];
    const float* gru_bih = (const float*)d_in[15];
    const float* gru_bhh = (const float*)d_in[16];
    const float* s2s_wih = (const float*)d_in[17];
    const float* s2s_whh = (const float*)d_in[18];
    const float* s2s_bih = (const float*)d_in[19];
    const float* s2s_bhh = (const float*)d_in[20];
    const float* mem_wih = (const float*)d_in[21];
    // d_in[22] = mem_whh (multiplies zero hidden state) -> unused
    const float* mem_bih = (const float*)d_in[23];
    const float* mem_bhh = (const float*)d_in[24];
    const float* lin1_w  = (const float*)d_in[25];
    const float* lin1_b  = (const float*)d_in[26];
    const float* lin2_w  = (const float*)d_in[27];
    const float* lin2_b  = (const float*)d_in[28];

    const int* src = edge_idx;
    const int* dst = edge_idx + N_EDGES;
    float* out = (float*)d_out;

    uintptr_t p = (uintptr_t)d_ws;
    float*    node_f32 = (float*)   carve(p, N_NODES * 64 * sizeof(float));
    float*    agg      = (float*)   carve(p, N_NODES * 64 * sizeof(float));
    float*    inv_deg  = (float*)   carve(p, N_NODES * sizeof(float));
    float*    e_arr    = (float*)   carve(p, N_NODES * sizeof(float));
    float*    hx_ws    = (float*)   carve(p, 64 * sizeof(float));
    _Float16* node_h   = (_Float16*)carve(p, N_NODES * 64 * sizeof(_Float16));
    _Float16* m_h      = (_Float16*)carve(p, N_NODES * 64 * sizeof(_Float16));
    _Float16* g_h      = (_Float16*)carve(p, N_EDGES * 64 * sizeof(_Float16));
    _Float16* w2f      = (_Float16*)carve(p, 262144 * sizeof(_Float16));
    _Float16* b2f      = (_Float16*)carve(p, 4096 * sizeof(_Float16));
    _Float16* wihf     = (_Float16*)carve(p, 12288 * sizeof(_Float16));
    _Float16* whhf     = (_Float16*)carve(p, 12288 * sizeof(_Float16));
    _Float16* lin1f    = (_Float16*)carve(p, 20480 * sizeof(_Float16));

    // degree -> inverse degree
    zero_f32<<<(N_NODES + 255) / 256, 256, 0, stream>>>(inv_deg, N_NODES);
    deg_kernel<<<(N_EDGES + 255) / 256, 256, 0, stream>>>(dst, inv_deg, N_EDGES);
    invdeg_kernel<<<(N_NODES + 255) / 256, 256, 0, stream>>>(inv_deg, N_NODES);

    // node init + edge hidden + fragment-packed weights
    init_node<<<N_NODES * 64 / 256, 256, 0, stream>>>(x, lin0_w, lin0_b, node_f32, node_h);
    edge_g<<<N_EDGES * 64 / 256, 256, 0, stream>>>(edge_attr, emlp_w1, emlp_b1, g_h);
    prep_w2r<<<262144 / 256, 256, 0, stream>>>(emlp_w2, w2f);
    prep_b2<<<4096 / 256, 256, 0, stream>>>(emlp_b2, b2f);
    prep_gruw<<<12288 / 256, 256, 0, stream>>>(gru_wih, wihf);
    prep_gruw<<<12288 / 256, 256, 0, stream>>>(gru_whh, whhf);
    prep_lin1<<<20480 / 256, 256, 0, stream>>>(lin1_w, lin1f);

    // 6 message-passing + GRU iterations
    for (int it = 0; it < 6; ++it) {
        zero_f32<<<N_NODES * 64 / 256, 256, 0, stream>>>(agg, N_NODES * 64);
        msg_kernel<<<(N_EDGES + 31) / 32, 32, 0, stream>>>(node_h, g_h, w2f, b2f,
                                                           src, dst, agg);
        m_kernel<<<N_NODES * 64 / 256, 256, 0, stream>>>(agg, inv_deg, node_f32,
                                                         conv_root, conv_bias, m_h);
        gru_kernel<<<N_NODES / 16, 32, 0, stream>>>(m_h, wihf, whhf, gru_bih, gru_bhh,
                                                    node_f32, node_h);
    }

    // Set2Set pooling + mem LSTM (writes hx/cx tail of d_out)
    set2set_kernel<<<1, 256, 0, stream>>>(node_f32, s2s_wih, s2s_whh, s2s_bih, s2s_bhh,
                                          mem_wih, mem_bih, mem_bhh,
                                          e_arr, hx_ws, out + T_ROWS * 6);

    // final MLP -> logits
    final_kernel<<<T_ROWS / 16, 32, 0, stream>>>(node_f32, nonring, hx_ws, lin1f,
                                                 lin1_b, lin2_w, lin2_b, out);
}
